// DynamicDecoupler_19877108645998
// MI455X (gfx1250) — matrix operand order repaired and checked by reading, not compiled, not run on hardware
//
#include <hip/hip_runtime.h>
#include <hip/hip_bf16.h>
#include <math.h>

// ---------------------------------------------------------------------------
// DynamicDecoupler for MI455X (gfx1250).
// Stage 1: avg-pool reduction (bandwidth trivial: 67 MB read)
// Stage 2: 16x64x72 GEMM via v_wmma_f32_16x16x32_f16 (one wave) + BN + softmax
// Stage 3: dynamic 3x3 reflect-pad stencil, float4 vectorized, writes low and
//          x-low in one pass (dominant: ~201 MB total -> ~8.6us @ 23.3 TB/s)
// ---------------------------------------------------------------------------

typedef __attribute__((ext_vector_type(16))) _Float16 v16h;
typedef __attribute__((ext_vector_type(8)))  float    v8f;

#define N_IMG 16
#define C_CH  64
#define H_DIM 128
#define W_DIM 128
#define GK2   72   // GROUP(8) * KSIZE^2(9)

// ---------------- Stage 1: per-(n,c) plane mean ----------------------------
__global__ void dd_pool(const float* __restrict__ x, float* __restrict__ pooled) {
    __shared__ float red[256];
    const int nc = blockIdx.x;                 // 0..1023
    const float4* plane = (const float4*)(x + (size_t)nc * (H_DIM * W_DIM));
    float s = 0.f;
    for (int i = threadIdx.x; i < (H_DIM * W_DIM) / 4; i += 256) {
        float4 v = plane[i];
        s += v.x + v.y + v.z + v.w;
    }
    red[threadIdx.x] = s;
    __syncthreads();
    for (int off = 128; off > 0; off >>= 1) {
        if (threadIdx.x < (unsigned)off) red[threadIdx.x] += red[threadIdx.x + off];
        __syncthreads();
    }
    if (threadIdx.x == 0) pooled[nc] = red[0] * (1.0f / (H_DIM * W_DIM));
}

// ---------------- Stage 2: WMMA GEMM + BN + softmax ------------------------
// f_raw[m][j] = sum_k pooled[m][k] * conv_w[j][k]   (m=0..15, j=0..71, k=0..63)
__global__ void dd_filter(const float* __restrict__ pooled,
                          const float* __restrict__ conv_w,
                          const float* __restrict__ gamma,
                          const float* __restrict__ beta,
                          const float* __restrict__ mean,
                          const float* __restrict__ var,
                          float* __restrict__ f_out) {
    __shared__ float fraw[N_IMG][GK2];

    const int tid = threadIdx.x;              // blockDim.x == 128 (4 waves)

    if (tid < 32) {                           // wave 0: EXEC all ones in-wave
        const int lane = tid;
        const int m    = lane & 15;           // A-matrix row
        const int hi   = lane >> 4;           // half-wave selector
        const int ncol = lane & 15;           // B/C column within tile

        // A: 16x64 f16, two 16x32 tiles. ISA 16-bit A layout (7.12.2):
        //   VGPR p (pair), bit b: K = 16*(p>=4) + 2*(p&3) + b + 8*hi
        v16h a0, a1;
#pragma unroll
        for (int e = 0; e < 16; ++e) {
            const int p = e >> 1, b = e & 1;
            const int k = ((p & 4) << 2) | ((p & 3) << 1) | b | (hi << 3);
            a0[e] = (_Float16)pooled[m * 64 + k];
            a1[e] = (_Float16)pooled[m * 64 + k + 32];
        }

        // 5 column tiles cover N=80 (>=72); lanes with nout>=72 feed zeros.
#pragma unroll
        for (int t = 0; t < 5; ++t) {
            const int nout = t * 16 + ncol;
            v16h b0 = {}, b1 = {};
            if (nout < GK2) {
                // B 32x16 f16 layout: K = 16*(e>=8) + 8*hi + (e&7), same as A
#pragma unroll
                for (int e = 0; e < 16; ++e) {
                    const int k = ((e >> 3) << 4) | (hi << 3) | (e & 7);
                    b0[e] = (_Float16)conv_w[nout * 64 + k];
                    b1[e] = (_Float16)conv_w[nout * 64 + k + 32];
                }
            }
            v8f cacc = {};
            cacc = __builtin_amdgcn_wmma_f32_16x16x32_f16(
                false, a0, false, b0, (short)0, cacc, false, false);
            cacc = __builtin_amdgcn_wmma_f32_16x16x32_f16(
                false, a1, false, b1, (short)0, cacc, false, false);
            if (nout < GK2) {
                // C/D layout: M = r + 8*hi, N = lane&15
#pragma unroll
                for (int r = 0; r < 8; ++r)
                    fraw[r + 8 * hi][nout] = cacc[r];
            }
        }
    }
    __syncthreads();

    // 128 threads = 16 images x 8 groups: BN (eval) + softmax over 9 taps
    const int n = tid >> 3;
    const int g = tid & 7;
    float v[9];
    float mx = -3.0e38f;
#pragma unroll
    for (int k = 0; k < 9; ++k) {
        const int j = g * 9 + k;
        const float fr = fraw[n][j];
        const float fb = gamma[j] * (fr - mean[j]) * rsqrtf(var[j] + 1e-5f) + beta[j];
        v[k] = fb;
        mx = fmaxf(mx, fb);
    }
    float s = 0.f;
#pragma unroll
    for (int k = 0; k < 9; ++k) { v[k] = __expf(v[k] - mx); s += v[k]; }
    const float inv = 1.0f / s;
#pragma unroll
    for (int k = 0; k < 9; ++k) f_out[tid * 9 + k] = v[k] * inv;
}

// ---------------- Stage 3: dynamic 3x3 reflect stencil ---------------------
// block (32,8): 32 lanes x 4 pixels = one 128-wide row; 8 rows per block.
// grid (H/8, C, N).
__global__ void dd_apply(const float* __restrict__ x,
                         const float* __restrict__ f,
                         float* __restrict__ low_out,
                         float* __restrict__ high_out) {
    const int n  = blockIdx.z;
    const int c  = blockIdx.y;
    const int h  = blockIdx.x * blockDim.y + threadIdx.y;
    const int w0 = threadIdx.x * 4;

    // filter: group = c/8 (repeat over 8 channels per group)
    const float* fw = f + (size_t)(n * 8 + (c >> 3)) * 9;
    float fk[9];
#pragma unroll
    for (int i = 0; i < 9; ++i) fk[i] = fw[i];

    const float* plane = x + ((size_t)(n * C_CH + c)) * (H_DIM * W_DIM);

    float acc0 = 0.f, acc1 = 0.f, acc2 = 0.f, acc3 = 0.f;
    float4 center;
#pragma unroll
    for (int dy = 0; dy < 3; ++dy) {
        int src = h + dy - 1;
        if (src < 0) src = -src;                        // reflect (no edge repeat)
        if (src >= H_DIM) src = 2 * H_DIM - 2 - src;
        const float* row = plane + src * W_DIM;

        const float4 mid = *(const float4*)(row + w0);  // 16B-aligned
        const float left  = (w0 == 0)            ? row[1]         : row[w0 - 1];
        const float right = (w0 + 4 >= W_DIM)    ? row[W_DIM - 2] : row[w0 + 4];
        if (dy == 1) center = mid;                       // reuse as x for high

        const float fL = fk[dy * 3 + 0];
        const float fC = fk[dy * 3 + 1];
        const float fR = fk[dy * 3 + 2];
        acc0 = fmaf(fL, left,  fmaf(fC, mid.x, fmaf(fR, mid.y, acc0)));
        acc1 = fmaf(fL, mid.x, fmaf(fC, mid.y, fmaf(fR, mid.z, acc1)));
        acc2 = fmaf(fL, mid.y, fmaf(fC, mid.z, fmaf(fR, mid.w, acc2)));
        acc3 = fmaf(fL, mid.z, fmaf(fC, mid.w, fmaf(fR, right, acc3)));
    }

    const size_t base = ((size_t)(n * C_CH + c)) * (H_DIM * W_DIM)
                      + (size_t)h * W_DIM + w0;
    *(float4*)(low_out  + base) = make_float4(acc0, acc1, acc2, acc3);
    *(float4*)(high_out + base) = make_float4(center.x - acc0, center.y - acc1,
                                              center.z - acc2, center.w - acc3);
}

// ---------------------------------------------------------------------------
extern "C" void kernel_launch(void* const* d_in, const int* in_sizes, int n_in,
                              void* d_out, int out_size, void* d_ws, size_t ws_size,
                              hipStream_t stream) {
    (void)in_sizes; (void)n_in; (void)out_size; (void)ws_size;

    const float* x      = (const float*)d_in[0];
    const float* conv_w = (const float*)d_in[1];
    const float* gamma  = (const float*)d_in[2];
    const float* beta   = (const float*)d_in[3];
    const float* mean   = (const float*)d_in[4];
    const float* var    = (const float*)d_in[5];

    float* out    = (float*)d_out;
    float* pooled = (float*)d_ws;          // 1024 floats
    float* fsm    = pooled + 1024;         // 16*72 = 1152 floats

    const size_t nchw = (size_t)N_IMG * C_CH * H_DIM * W_DIM;

    dd_pool<<<N_IMG * C_CH, 256, 0, stream>>>(x, pooled);
    dd_filter<<<1, 128, 0, stream>>>(pooled, conv_w, gamma, beta, mean, var, fsm);

    dim3 grid(H_DIM / 8, C_CH, N_IMG);
    dim3 block(32, 8);
    dd_apply<<<grid, block, 0, stream>>>(x, fsm, out, out + nchw);
}
